// DecodeState_23811298689268
// MI455X (gfx1250) — 8-bit matrix operand order repaired and checked by reading, compiled, not run on hardware
//
#include <hip/hip_runtime.h>
#include <hip/hip_bf16.h>


typedef __attribute__((ext_vector_type(8))) int v8i;

constexpr int Qc  = 4194304;
constexpr int MTc = 1048576;
constexpr int MSc = 4096;
constexpr int NB  = MSc + 1;

constexpr int SCHUNK = 2048;
constexpr int SBLK   = MTc / SCHUNK;
constexpr int PCHUNK = 2048;
constexpr int PBLK   = Qc / PCHUNK;

static_assert(SBLK * SCHUNK == MTc, "sort tiling");
static_assert(PBLK * PCHUNK == Qc, "purge tiling");
static_assert(SBLK % 16 == 0, "wmma row tiles");

constexpr size_t OFF_PT  = 0;
constexpr size_t OFF_PS  = (size_t)MTc;
constexpr size_t OFF_PP  = (size_t)2 * MTc;
constexpr size_t OFF_PL  = (size_t)3 * MTc;
constexpr size_t OFF_NUM = (size_t)4 * MTc;
constexpr size_t OFF_CNT = OFF_NUM + 1;
constexpr size_t OFF_QT  = OFF_CNT + MSc;
constexpr size_t OFF_QS  = OFF_QT + Qc;
constexpr size_t OFF_QP  = OFF_QS + Qc;
constexpr size_t OFF_QL  = OFF_QP + Qc;
constexpr size_t OFF_NQ  = OFF_QL + Qc;

__device__ __forceinline__ unsigned ballot32(bool p) {
#if __has_builtin(__builtin_amdgcn_ballot_w32)
  return __builtin_amdgcn_ballot_w32(p);
#else
  return (unsigned)__ballot(p);
#endif
}

__device__ __forceinline__ int comb_i(const int* __restrict__ q,
                                      const int* __restrict__ n,
                                      int i, int nq0, int nn) {
  int rel = i - nq0;
  return (rel >= 0 && rel < nn) ? n[rel] : q[i];
}
__device__ __forceinline__ float comb_f(const float* __restrict__ q,
                                        const float* __restrict__ n,
                                        int i, int nq0, int nn) {
  int rel = i - nq0;
  return (rel >= 0 && rel < nn) ? n[rel] : q[i];
}

__global__ __launch_bounds__(256) void k_fill(float* __restrict__ out) {
  int i = blockIdx.x * 256 + threadIdx.x;
  out[OFF_QT + i] = -1.0f;
  out[OFF_QS + i] = -1.0f;
  out[OFF_QP + i] = -1.0f;
  out[OFF_QL + i] = 0.0f;
}

__global__ __launch_bounds__(256) void k_sort_hist(const int* __restrict__ qs,
                                                   const int* __restrict__ ns,
                                                   const int* __restrict__ pNq0,
                                                   const int* __restrict__ pNn,
                                                   int* __restrict__ blockHist) {
  __shared__ int h[NB];
  const int t = threadIdx.x, b = blockIdx.x;
  for (int v = t; v < NB; v += 256) h[v] = 0;
  __syncthreads();
  const int nq0 = pNq0[0], nn = pNn[0];
  const int base = b * SCHUNK;
  for (int r = 0; r < SCHUNK / 256; ++r) {
    int i = base + r * 256 + t;
    __builtin_prefetch(&qs[min(i + SCHUNK, Qc - 1)], 0, 1);
    int s = comb_i(qs, ns, i, nq0, nn);
    int bin = min(max(s + 1, 0), NB - 1);
    atomicAdd(&h[bin], 1);
  }
  __syncthreads();
  for (int v = t; v < NB; v += 256) blockHist[(size_t)b * NB + v] = h[v];
}

__global__ __launch_bounds__(256) void k_wmma_scan(int* __restrict__ blockHist,
                                                   int* __restrict__ binTotal) {
  __shared__ int Xs[SBLK * 16];
  const int ct = blockIdx.x;
  const int v0 = ct * 16;
  const int t = threadIdx.x;
  const int cvalid = min(16, NB - v0);

  {
    const int c = t & 15;
    const int r0 = t >> 4;
    for (int rb = 0; rb < SBLK; rb += 16) {
      int b = rb + r0;
      int x = (c < cvalid) ? blockHist[(size_t)b * NB + v0 + c] : 0;
      Xs[b * 16 + c] = x;
    }
  }
  __syncthreads();

  if (t < cvalid) {
    int s = 0;
    for (int b = 0; b < SBLK; ++b) s += Xs[b * 16 + t];
    binTotal[v0 + t] = s;
  }

  const int wave = __builtin_amdgcn_readfirstlane(t >> 5);
  const int lane = t & 31;
  const int g = lane >> 4;
  const int m = lane & 15;

  for (int rtp = 0; rtp < 4; ++rtp) {
    const int rt = wave + rtp * 8;
    const int rowBase = rt * 16;
    v8i acc0 = {0, 0, 0, 0, 0, 0, 0, 0};
    v8i acc1 = {0, 0, 0, 0, 0, 0, 0, 0};
    for (int kb = 0; kb * 64 < rowBase + 16; ++kb) {
      v8i A, B0, B1;
      for (int v = 0; v < 8; ++v) {
        int Ka = ((v & 2) ? 16 : 0) + ((v & 1) ? 4 : 0) + g * 8 + ((v >> 2) * 32) + kb * 64;
        unsigned pa = 0;
#pragma unroll
        for (int by = 0; by < 4; ++by) {
          int K = Ka + by;
          if (K < rowBase + m) pa |= (1u << (8 * by));
        }
        A[v] = (int)pa;
        int Kb = ((v & 2) ? 16 : 0) + ((v & 1) ? 4 : 0) + g * 8 + ((v >> 2) * 32) + kb * 64;
        unsigned p0 = 0, p1 = 0;
#pragma unroll
        for (int by = 0; by < 4; ++by) {
          int x = Xs[(Kb + by) * 16 + m];
          p0 |= (unsigned)(x & 0xFF) << (8 * by);
          p1 |= (unsigned)((x >> 8) & 0xFF) << (8 * by);
        }
        B0[v] = (int)p0;
        B1[v] = (int)p1;
      }
      acc0 = __builtin_amdgcn_wmma_i32_16x16x64_iu8(false, A, false, B0, acc0, false, false);
      acc1 = __builtin_amdgcn_wmma_i32_16x16x64_iu8(false, A, false, B1, acc1, false, false);
    }
    if (m < cvalid) {
#pragma unroll
      for (int r = 0; r < 8; ++r) {
        int row = rowBase + r + g * 8;
        int y = acc0[r] + (acc1[r] << 8);
        blockHist[(size_t)row * NB + v0 + m] = y;
      }
    }
  }
}

__global__ __launch_bounds__(256) void k_bin_scan(const int* __restrict__ binTotal,
                                                  int* __restrict__ binStart,
                                                  float* __restrict__ out,
                                                  const int* __restrict__ pNq0,
                                                  const int* __restrict__ pNn) {
  __shared__ int lstart[NB];
  __shared__ int wsum[8];
  const int t = threadIdx.x, lane = t & 31, wave = t >> 5;
  const int v0 = t * 17;
  const int cnt = max(0, min(17, NB - v0));
  int s = 0;
  for (int k = 0; k < cnt; ++k) s += binTotal[v0 + k];
  int incl = s;
  for (int d = 1; d < 32; d <<= 1) {
    int y = __shfl_up(incl, d, 32);
    if (lane >= d) incl += y;
  }
  if (lane == 31) wsum[wave] = incl;
  __syncthreads();
  int wofs = 0;
  for (int w = 0; w < wave; ++w) wofs += wsum[w];
  int excl = wofs + incl - s;
  for (int k = 0; k < cnt; ++k) {
    lstart[v0 + k] = excl;
    binStart[v0 + k] = excl;
    excl += binTotal[v0 + k];
  }
  __syncthreads();
  const int num = min(pNq0[0] + pNn[0], MTc);
  if (t == 0) out[OFF_NUM] = (float)num;
  for (int v = t; v < MSc; v += 256) {
    int st = lstart[v + 1];
    int tot = binTotal[v + 1];
    int c = min(max(num - st, 0), tot);
    if (v == 0) c += min(max(num - lstart[0], 0), binTotal[0]);
    out[OFF_CNT + v] = (float)c;
  }
}

__global__ __launch_bounds__(32) void k_sort_scatter(
    const int* __restrict__ qt, const int* __restrict__ qs,
    const int* __restrict__ qp, const float* __restrict__ ql,
    const int* __restrict__ nt, const int* __restrict__ ns,
    const int* __restrict__ np, const float* __restrict__ nl,
    const int* __restrict__ blockScan, const int* __restrict__ binStart,
    float* __restrict__ out, const int* __restrict__ pNq0,
    const int* __restrict__ pNn) {
  __shared__ int hist[NB];
  const int b = blockIdx.x, lane = threadIdx.x;
  const int nq0 = pNq0[0], nn = pNn[0];
  for (int v = lane; v < NB; v += 32)
    hist[v] = binStart[v] + blockScan[(size_t)b * NB + v];
  __syncthreads();
  float* PT = out + OFF_PT;
  float* PS = out + OFF_PS;
  float* PP = out + OFF_PP;
  float* PL = out + OFF_PL;
  const int base = b * SCHUNK;
  for (int r = 0; r < SCHUNK / 32; ++r) {
    int i = base + r * 32 + lane;
    int s = comb_i(qs, ns, i, nq0, nn);
    int bin = min(max(s + 1, 0), NB - 1);
    unsigned mask = 0;
    for (int l = 0; l < 32; ++l) {
      int ob = __shfl(bin, l, 32);
      mask |= (ob == bin ? 1u : 0u) << l;
    }
    int rank = __popc(mask & ((1u << lane) - 1u));
    int leader = __ffs(mask) - 1;
    int cnt = __popc(mask);
    int baseOfs = 0;
    if (lane == leader) baseOfs = atomicAdd(&hist[bin], cnt);
    baseOfs = __shfl(baseOfs, leader, 32);
    int dest = baseOfs + rank;
    PT[dest] = (float)comb_i(qt, nt, i, nq0, nn);
    PS[dest] = (float)s;
    PP[dest] = (float)comb_i(qp, np, i, nq0, nn);
    PL[dest] = comb_f(ql, nl, i, nq0, nn);
  }
}

__global__ __launch_bounds__(256) void k_purge_count(
    const int* __restrict__ qs, const int* __restrict__ ns,
    const int* __restrict__ pNq0, const int* __restrict__ pNn,
    const int* __restrict__ pP, int* __restrict__ blockKeep,
    int* __restrict__ blockValid) {
  __shared__ int sk[256], sv[256];
  const int t = threadIdx.x, b = blockIdx.x;
  const int nq0 = pNq0[0], nn = pNn[0], P = pP[0];
  const int num = min(nq0 + nn, MTc);
  const int limit = Qc - num;
  int kc = 0, vc = 0;
  const int base = b * PCHUNK;
  for (int r = 0; r < PCHUNK / 256; ++r) {
    int i = base + r * 256 + t;
    int j = i + num;
    __builtin_prefetch(&qs[min(j + PCHUNK, Qc - 1)], 0, 1);
    int s = (i < limit) ? comb_i(qs, ns, j, nq0, nn) : -1;
    int keep = (s != P) ? 1 : 0;
    kc += keep;
    vc += (keep && s != -1) ? 1 : 0;
  }
  sk[t] = kc;
  sv[t] = vc;
  __syncthreads();
  for (int o = 128; o > 0; o >>= 1) {
    if (t < o) { sk[t] += sk[t + o]; sv[t] += sv[t + o]; }
    __syncthreads();
  }
  if (t == 0) { blockKeep[b] = sk[0]; blockValid[b] = sv[0]; }
}

__global__ __launch_bounds__(256) void k_purge_scan(
    const int* __restrict__ blockKeep, const int* __restrict__ blockValid,
    int* __restrict__ keepPrefix, float* __restrict__ out) {
  __shared__ int wsum[8], wval[8];
  const int t = threadIdx.x, lane = t & 31, wave = t >> 5;
  const int base = t * (PBLK / 256);
  int s = 0, v = 0;
  for (int k = 0; k < PBLK / 256; ++k) {
    s += blockKeep[base + k];
    v += blockValid[base + k];
  }
  int incl = s;
  for (int d = 1; d < 32; d <<= 1) {
    int y = __shfl_up(incl, d, 32);
    if (lane >= d) incl += y;
  }
  if (lane == 31) wsum[wave] = incl;
  for (int d = 16; d > 0; d >>= 1) v += __shfl_xor(v, d, 32);
  if (lane == 0) wval[wave] = v;
  __syncthreads();
  int wofs = 0;
  for (int w = 0; w < wave; ++w) wofs += wsum[w];
  int excl = wofs + incl - s;
  for (int k = 0; k < PBLK / 256; ++k) {
    keepPrefix[base + k] = excl;
    excl += blockKeep[base + k];
  }
  if (t == 0) {
    int tv = 0;
    for (int w = 0; w < 8; ++w) tv += wval[w];
    out[OFF_NQ] = (float)tv;
  }
}

__global__ __launch_bounds__(32) void k_purge_scatter(
    const int* __restrict__ qt, const int* __restrict__ qs,
    const int* __restrict__ qp, const float* __restrict__ ql,
    const int* __restrict__ nt, const int* __restrict__ ns,
    const int* __restrict__ np, const float* __restrict__ nl,
    const int* __restrict__ keepPrefix, float* __restrict__ out,
    const int* __restrict__ pNq0, const int* __restrict__ pNn,
    const int* __restrict__ pP) {
  const int b = blockIdx.x, lane = threadIdx.x;
  const int nq0 = pNq0[0], nn = pNn[0], P = pP[0];
  const int num = min(nq0 + nn, MTc);
  const int limit = Qc - num;
  float* QT = out + OFF_QT;
  float* QS = out + OFF_QS;
  float* QP = out + OFF_QP;
  float* QL = out + OFF_QL;
  int runBase = keepPrefix[b];
  const int base = b * PCHUNK;
  for (int r = 0; r < PCHUNK / 32; ++r) {
    int i = base + r * 32 + lane;
    int j = i + num;
    bool inBuf = i < limit;
    int s = inBuf ? comb_i(qs, ns, j, nq0, nn) : -1;
    bool keep = (s != P);
    unsigned bal = ballot32(keep);
    int rank = __popc(bal & ((1u << lane) - 1u));
    if (keep) {
      int dest = runBase + rank;
      int tv = inBuf ? comb_i(qt, nt, j, nq0, nn) : -1;
      int pv = inBuf ? comb_i(qp, np, j, nq0, nn) : -1;
      float lv = inBuf ? comb_f(ql, nl, j, nq0, nn) : 0.0f;
      QT[dest] = (float)tv;
      QS[dest] = (float)s;
      QP[dest] = (float)pv;
      QL[dest] = lv;
    }
    runBase += __popc(bal);
  }
}

extern "C" void kernel_launch(void* const* d_in, const int* in_sizes, int n_in,
                              void* d_out, int out_size, void* d_ws, size_t ws_size,
                              hipStream_t stream) {
  const float* qL = (const float*)d_in[0];
  const float* nL = (const float*)d_in[1];
  const int* qT = (const int*)d_in[2];
  const int* qS = (const int*)d_in[3];
  const int* qP = (const int*)d_in[4];
  const int* nT = (const int*)d_in[5];
  const int* nS = (const int*)d_in[6];
  const int* nP = (const int*)d_in[7];
  const int* pNq0 = (const int*)d_in[8];
  const int* pNn = (const int*)d_in[9];
  const int* pPurge = (const int*)d_in[11];
  float* out = (float*)d_out;

  char* ws = (char*)d_ws;
  int* blockHist = (int*)ws;  ws += (size_t)SBLK * NB * sizeof(int);
  int* binTotal  = (int*)ws;  ws += (size_t)NB * sizeof(int);
  int* binStart  = (int*)ws;  ws += (size_t)NB * sizeof(int);
  int* blockKeep = (int*)ws;  ws += (size_t)PBLK * sizeof(int);
  int* blockValid= (int*)ws;  ws += (size_t)PBLK * sizeof(int);
  int* keepPref  = (int*)ws;  ws += (size_t)PBLK * sizeof(int);

  k_fill<<<Qc / 256, 256, 0, stream>>>(out);
  k_sort_hist<<<SBLK, 256, 0, stream>>>(qS, nS, pNq0, pNn, blockHist);
  k_wmma_scan<<<(NB + 15) / 16, 256, 0, stream>>>(blockHist, binTotal);
  k_bin_scan<<<1, 256, 0, stream>>>(binTotal, binStart, out, pNq0, pNn);
  k_sort_scatter<<<SBLK, 32, 0, stream>>>(qT, qS, qP, qL, nT, nS, nP, nL,
                                          blockHist, binStart, out, pNq0, pNn);
  k_purge_count<<<PBLK, 256, 0, stream>>>(qS, nS, pNq0, pNn, pPurge,
                                          blockKeep, blockValid);
  k_purge_scan<<<1, 256, 0, stream>>>(blockKeep, blockValid, keepPref, out);
  k_purge_scatter<<<PBLK, 32, 0, stream>>>(qT, qS, qP, qL, nT, nS, nP, nL,
                                           keepPref, out, pNq0, pNn, pPurge);
}
